// DeepNT_1717986918869
// MI455X (gfx1250) — compile-verified, matrix operand order checked by reading, not run on hardware
//
#include <hip/hip_runtime.h>
#include <hip/hip_bf16.h>
#include <stdint.h>

// ---------------------------------------------------------------------------
// Types for WMMA (wave32, gfx1250)
// ---------------------------------------------------------------------------
typedef __attribute__((ext_vector_type(16))) __bf16 v16bf;
typedef __attribute__((ext_vector_type(8)))  __bf16 v8bf;
typedef __attribute__((ext_vector_type(8)))  float  v8f;
typedef __attribute__((ext_vector_type(4)))  float  f4;

static __device__ __forceinline__ __bf16 f32_to_bf16_rne(float f) {
  union { float f; uint32_t u; } v; v.f = f;
  uint32_t r = v.u + 0x7FFFu + ((v.u >> 16) & 1u);   // round-to-nearest-even
  uint16_t h = (uint16_t)(r >> 16);
  __bf16 out;
  __builtin_memcpy(&out, &h, sizeof(out));
  return out;
}

// ---------------------------------------------------------------------------
// fp32 -> bf16 conversion (grid-stride-free: exact grids, n % 1024 == 0)
// ---------------------------------------------------------------------------
__launch_bounds__(256)
__global__ void cvt_f32_bf16(const float* __restrict__ in,
                             __bf16* __restrict__ out, int n)
{
  size_t i = ((size_t)blockIdx.x * 256 + threadIdx.x) * 4;
  f4 v = *reinterpret_cast<const f4*>(in + i);
  out[i + 0] = f32_to_bf16_rne(v[0]);
  out[i + 1] = f32_to_bf16_rne(v[1]);
  out[i + 2] = f32_to_bf16_rne(v[2]);
  out[i + 3] = f32_to_bf16_rne(v[3]);
  (void)n;
}

// ---------------------------------------------------------------------------
// bf16-WMMA GEMM:  C[M,N] = (relu?) A[M,K] * B[K,N]   (A,B bf16 in HBM)
//   - A tile (128x32) staged via GLOBAL_LOAD_ASYNC_TO_LDS_B128 (ASYNCcnt)
//   - B tile (32x128) staged transposed via regular loads (needs transpose)
//   - 8 waves, wave tile 64x32 -> 4x2 fragments, K step 32
//   - output f32 or bf16
// ---------------------------------------------------------------------------
#define BM 128
#define BN 128
#define BK 32
#define LDSS 40   // bf16 elements per LDS row: 80 bytes, 16B-aligned halves

__launch_bounds__(256)
__global__ void wmma_gemm_bf16_async(const __bf16* __restrict__ A,
                                     const __bf16* __restrict__ B,
                                     void* __restrict__ Cout,
                                     int M, int N, int K,
                                     int do_relu, int out_bf16)
{
  __shared__ __align__(128) unsigned char smem[2 * BM * LDSS * 2];
  __bf16 (*As)[LDSS]  = reinterpret_cast<__bf16(*)[LDSS]>(smem);
  __bf16 (*BsT)[LDSS] = reinterpret_cast<__bf16(*)[LDSS]>(smem + (size_t)BM * LDSS * 2);

  const int tid  = threadIdx.x;
  const int lane = tid & 31;
  const int wave = tid >> 5;
  const int wm   = wave >> 2;          // 0..1  (64-row slab)
  const int wn   = wave & 3;           // 0..3  (32-col slab)

  const int m0 = blockIdx.x * BM;
  const int n0 = blockIdx.y * BN;

  v8f vzero = {};
  v8f acc[4][2];
  for (int i = 0; i < 4; ++i)
    for (int j = 0; j < 2; ++j)
      acc[i][j] = vzero;

  const int fidx = lane & 15;          // M (A) / N (B) index within fragment
  const int ksel = (lane >> 4) << 3;   // 0 or 8: K sub-block select (ISA layout)

  for (int kb = 0; kb < K; kb += BK) {
    // ---- A tile: async copy HBM -> LDS, 512 x 16B chunks, 2 per thread ----
    for (int it = 0; it < 2; ++it) {
      int idx = tid + it * 256;                 // 0..511
      int r   = idx >> 2;                       // 0..127 (row)
      int c   = idx & 3;                        // 0..3   (16B chunk in row)
      const __bf16* gp = &A[(size_t)(m0 + r) * K + kb + c * 8];
      uint32_t ldsoff = (uint32_t)(uintptr_t)(const void*)&As[r][c * 8];
      asm volatile("global_load_async_to_lds_b128 %0, %1, off"
                   :: "v"(ldsoff), "v"((unsigned long long)(uintptr_t)gp)
                   : "memory");
    }

    // ---- B tile: load row-major bf16, store transposed into LDS ----
    for (int it = 0; it < 2; ++it) {
      int idx = tid + it * 256;                 // 0..511
      int kr  = idx >> 4;                       // 0..31 (k row)
      int c   = idx & 15;                       // 0..15 (chunk of 8 cols)
      v8bf bv = *reinterpret_cast<const v8bf*>(&B[(size_t)(kb + kr) * N + n0 + c * 8]);
      for (int j = 0; j < 8; ++j)
        BsT[c * 8 + j][kr] = bv[j];
    }

    // prefetch next B slab (emits global_prefetch_b8)
    if (kb + BK < K)
      __builtin_prefetch(&B[(size_t)(kb + BK + (tid & 31)) * N + n0], 0, 1);

    // wait for this wave's async copies, then workgroup barrier
    asm volatile("s_wait_asynccnt 0x0" ::: "memory");
    __syncthreads();

    // ---- load fragments (two ds_load_b128 each) ----
    v16bf afrag[4], bfrag[2];
    for (int mi = 0; mi < 4; ++mi) {
      const __bf16* p = &As[wm * 64 + mi * 16 + fidx][ksel];
      v8bf lo = *reinterpret_cast<const v8bf*>(p);        // K = ksel+0..7
      v8bf hi = *reinterpret_cast<const v8bf*>(p + 16);   // K = 16+ksel+0..7
      afrag[mi] = __builtin_shufflevector(lo, hi,
          0,1,2,3,4,5,6,7,8,9,10,11,12,13,14,15);
    }
    for (int ni = 0; ni < 2; ++ni) {
      const __bf16* p = &BsT[wn * 32 + ni * 16 + fidx][ksel];
      v8bf lo = *reinterpret_cast<const v8bf*>(p);
      v8bf hi = *reinterpret_cast<const v8bf*>(p + 16);
      bfrag[ni] = __builtin_shufflevector(lo, hi,
          0,1,2,3,4,5,6,7,8,9,10,11,12,13,14,15);
    }

    // ---- 8 WMMAs per wave per K-step ----
    for (int mi = 0; mi < 4; ++mi)
      for (int ni = 0; ni < 2; ++ni)
        acc[mi][ni] = __builtin_amdgcn_wmma_f32_16x16x32_bf16(
            /*neg_a=*/false, afrag[mi],
            /*neg_b=*/false, bfrag[ni],
            /*c_mod=*/(short)0, acc[mi][ni],
            /*reuse_a=*/false, /*reuse_b=*/false);

    __syncthreads();
  }

  // ---- epilogue: C/D ISA layout (row = e + 8*(lane>=16), col = lane&15) ----
  const int rhi = (lane >> 4) * 8;
  for (int mi = 0; mi < 4; ++mi)
    for (int ni = 0; ni < 2; ++ni) {
      const int col   = n0 + wn * 32 + ni * 16 + fidx;
      const int rbase = m0 + wm * 64 + mi * 16 + rhi;
      for (int e = 0; e < 8; ++e) {
        float val = acc[mi][ni][e];
        if (do_relu) val = fmaxf(val, 0.0f);
        if (out_bf16)
          reinterpret_cast<__bf16*>(Cout)[(size_t)(rbase + e) * N + col] =
              f32_to_bf16_rne(val);
        else
          reinterpret_cast<float*>(Cout)[(size_t)(rbase + e) * N + col] = val;
      }
    }
}

// ---------------------------------------------------------------------------
// Fused: gather hu/hv, 3 rounds of residual path attention, final FC.
// One 128-thread block per batch element; d = feature index.
// ---------------------------------------------------------------------------
__launch_bounds__(128)
__global__ void path_attn_fc(const float* __restrict__ emb,
                             const int* __restrict__ u,
                             const int* __restrict__ v,
                             const int* __restrict__ paths,
                             const float* __restrict__ Wq,
                             const float* __restrict__ Wfc,
                             const float* __restrict__ bfc,
                             float* __restrict__ out)
{
  const int b = blockIdx.x;
  const int d = threadIdx.x;             // 0..127

  __shared__ float q[2][128];
  __shared__ float kk[128];
  __shared__ float sc[16];
  __shared__ float red[128];

  q[0][d] = emb[(size_t)u[b] * 128 + d];
  q[1][d] = emb[(size_t)v[b] * 128 + d];
  __syncthreads();

  const float inv_sqrt_d = 0.08838834764831845f;  // 1/sqrt(128)

  for (int p = 0; p < 3; ++p) {
    for (int w = 0; w < 2; ++w) {
      // k = q @ Wq   (k[d] = sum_j q[j] * Wq[j,d])
      float a = 0.0f;
      for (int j = 0; j < 128; ++j) a += q[w][j] * Wq[j * 128 + d];
      kk[d] = a;
      __syncthreads();

      // scores over the 10 path nodes
      if (d < 10) {
        int node = paths[(size_t)b * 30 + p * 10 + d];
        const float* pe = &emb[(size_t)node * 128];
        float s = 0.0f;
        for (int j = 0; j < 128; ++j) s += pe[j] * kk[j];
        sc[d] = s * inv_sqrt_d;
      }
      __syncthreads();

      // softmax (computed redundantly per lane) + weighted path sum
      float mx = sc[0];
      for (int l = 1; l < 10; ++l) mx = fmaxf(mx, sc[l]);
      float wts[10], sum = 0.0f;
      for (int l = 0; l < 10; ++l) { wts[l] = __expf(sc[l] - mx); sum += wts[l]; }
      const float inv = 1.0f / sum;
      float upd = 0.0f;
      for (int l = 0; l < 10; ++l) {
        int node = paths[(size_t)b * 30 + p * 10 + l];
        upd += wts[l] * inv * emb[(size_t)node * 128 + d];
      }
      __syncthreads();
      q[w][d] += upd;                    // residual
      __syncthreads();
    }
  }

  // out[b] = concat(hu,hv) @ Wfc + bfc
  red[d] = q[0][d] * Wfc[d] + q[1][d] * Wfc[128 + d];
  __syncthreads();
  for (int off = 64; off > 0; off >>= 1) {
    if (d < off) red[d] += red[d + off];
    __syncthreads();
  }
  if (d == 0) out[b] = red[0] + bfc[0];
}

// ---------------------------------------------------------------------------
// Host-side launch sequence
// ---------------------------------------------------------------------------
extern "C" void kernel_launch(void* const* d_in, const int* in_sizes, int n_in,
                              void* d_out, int out_size, void* d_ws, size_t ws_size,
                              hipStream_t stream)
{
  (void)in_sizes; (void)n_in; (void)out_size; (void)ws_size;

  const float* x     = (const float*)d_in[0];
  const int*   u     = (const int*)  d_in[1];
  const int*   v     = (const int*)  d_in[2];
  const float* adj   = (const float*)d_in[3];
  const int*   paths = (const int*)  d_in[4];
  const float* W0    = (const float*)d_in[5];
  const float* W1    = (const float*)d_in[6];
  const float* W2    = (const float*)d_in[7];
  const float* Wq    = (const float*)d_in[8];
  const float* Wfc   = (const float*)d_in[9];
  const float* bfc   = (const float*)d_in[10];
  float* out = (float*)d_out;

  // ---- workspace layout (bytes; ~151 MB total) ----
  char* ws = (char*)d_ws;
  __bf16* adjbf = (__bf16*)ws;                              // 8192*8192 bf16 = 128 MB
  __bf16* xbf   = (__bf16*)(ws + (size_t)134217728);        // 8192*256  bf16 =   4 MB
  __bf16* w0bf  = (__bf16*)(ws + (size_t)138412032);        // 256*256
  __bf16* w1bf  = (__bf16*)(ws + (size_t)138543104);        // 256*256
  __bf16* w2bf  = (__bf16*)(ws + (size_t)138674176);        // 256*128
  __bf16* bufA  = (__bf16*)(ws + (size_t)138739712);        // 8192*256 bf16
  __bf16* bufB  = (__bf16*)(ws + (size_t)142934016);        // 8192*256 bf16
  float*  emb   = (float*) (ws + (size_t)147128320);        // 8192*128 f32

  const dim3 blk(256);

  // ---- precision downconvert: adj/x/W* -> bf16 (one pass; adj now fits L2) ----
  cvt_f32_bf16<<<dim3(65536), blk, 0, stream>>>(adj, adjbf, 8192 * 8192);
  cvt_f32_bf16<<<dim3(2048),  blk, 0, stream>>>(x,   xbf,   8192 * 256);
  cvt_f32_bf16<<<dim3(64),    blk, 0, stream>>>(W0,  w0bf,  256 * 256);
  cvt_f32_bf16<<<dim3(64),    blk, 0, stream>>>(W1,  w1bf,  256 * 256);
  cvt_f32_bf16<<<dim3(32),    blk, 0, stream>>>(W2,  w2bf,  256 * 128);

  // T0 = x @ W0                          [8192,256]x[256,256]
  wmma_gemm_bf16_async<<<dim3(64, 2), blk, 0, stream>>>(xbf,   w0bf, bufA, 8192, 256, 256,  0, 1);
  // H1 = relu(adj @ T0)                  [8192,8192]x[8192,256]
  wmma_gemm_bf16_async<<<dim3(64, 2), blk, 0, stream>>>(adjbf, bufA, bufB, 8192, 256, 8192, 1, 1);
  // T1 = H1 @ W1
  wmma_gemm_bf16_async<<<dim3(64, 2), blk, 0, stream>>>(bufB,  w1bf, bufA, 8192, 256, 256,  0, 1);
  // H2 = relu(adj @ T1)
  wmma_gemm_bf16_async<<<dim3(64, 2), blk, 0, stream>>>(adjbf, bufA, bufB, 8192, 256, 8192, 1, 1);
  // T2 = H2 @ W2                         [8192,256]x[256,128]
  wmma_gemm_bf16_async<<<dim3(64, 1), blk, 0, stream>>>(bufB,  w2bf, bufA, 8192, 128, 256,  0, 1);
  // EMB = adj @ T2  (no relu, f32 out)   [8192,8192]x[8192,128]
  wmma_gemm_bf16_async<<<dim3(64, 1), blk, 0, stream>>>(adjbf, bufA, emb,  8192, 128, 8192, 0, 0);

  // gather + 3x residual path attention + final FC
  path_attn_fc<<<dim3(4096), dim3(128), 0, stream>>>(emb, u, v, paths, Wq, Wfc, bfc, out);
}
